// MultiheadAttention_53549652246692
// MI455X (gfx1250) — matrix operand order repaired and checked by reading, not compiled, not run on hardware
//
#include <hip/hip_runtime.h>
#include <hip/hip_bf16.h>

typedef __bf16 bf16;
typedef __attribute__((ext_vector_type(16))) __bf16 v16bf;
typedef __attribute__((ext_vector_type(8)))  float  v8f;

union FragBF { v16bf v; uint4 q[2]; };
union Pack4  { uint2 u; bf16 h[4]; };
union Pack8  { uint4 u; bf16 h[8]; };

#define D_MODEL 1024
#define N_HEADS 16
#define D_HEAD  64
#define BATCH   4
#define SEQ     2048
#define M_TOK   (BATCH * SEQ)   // 8192

static __device__ __forceinline__ v8f wmma_bf16(const FragBF& a, const FragBF& b, v8f c) {
  return __builtin_amdgcn_wmma_f32_16x16x32_bf16(false, a.v, false, b.v, (short)0, c, false, false);
}

// ---------------------------------------------------------------------------
// GEMM: Y[m,n] = sum_k X[m,k] * W[n,k] + bias[n]
// mode 0: store bf16, head-major layout [B, H, S, 64]  (for Q/K/V projections)
// mode 1: store f32 row-major [M, N]                   (for output projection)
// Block tile 128x64, 8 waves, each wave 32x32 (2x2 wmma tiles), K step 32.
// ---------------------------------------------------------------------------
__global__ __launch_bounds__(256)
void gemm_xwt_kernel(const float* __restrict__ X, const float* __restrict__ W,
                     const float* __restrict__ bias, bf16* __restrict__ outH,
                     float* __restrict__ outF, int mode) {
  __shared__ bf16 Xs[128][40];
  __shared__ bf16 Ws[64][40];

  const int tid  = threadIdx.x;
  const int wave = tid >> 5, lane = tid & 31;
  const int n16  = lane & 15, hi = lane >> 4;
  const int m0   = blockIdx.y * 128;
  const int n0   = blockIdx.x * 64;
  const int mr   = (wave >> 1) * 32;
  const int nr   = (wave & 1) * 32;

  v8f acc[2][2] = {};

  for (int kt = 0; kt < D_MODEL / 32; ++kt) {
    const int k0 = kt * 32;
    // stage X tile 128x32 (f32 -> bf16)
    #pragma unroll
    for (int it = 0; it < 4; ++it) {
      int f4  = tid + 256 * it;
      int row = f4 >> 3, c4 = (f4 & 7) * 4;
      float4 xv = *reinterpret_cast<const float4*>(X + (size_t)(m0 + row) * D_MODEL + k0 + c4);
      Pack4 p; p.h[0] = (bf16)xv.x; p.h[1] = (bf16)xv.y; p.h[2] = (bf16)xv.z; p.h[3] = (bf16)xv.w;
      *reinterpret_cast<uint2*>(&Xs[row][c4]) = p.u;
    }
    // stage W tile 64x32 (f32 -> bf16)
    #pragma unroll
    for (int it = 0; it < 2; ++it) {
      int f4  = tid + 256 * it;
      int row = f4 >> 3, c4 = (f4 & 7) * 4;
      float4 wv = *reinterpret_cast<const float4*>(W + (size_t)(n0 + row) * D_MODEL + k0 + c4);
      Pack4 p; p.h[0] = (bf16)wv.x; p.h[1] = (bf16)wv.y; p.h[2] = (bf16)wv.z; p.h[3] = (bf16)wv.w;
      *reinterpret_cast<uint2*>(&Ws[row][c4]) = p.u;
    }
    __syncthreads();

    FragBF a[2], b[2];
    #pragma unroll
    for (int mi = 0; mi < 2; ++mi) {
      const bf16* rp = &Xs[mr + mi * 16 + n16][0];
      a[mi].q[0] = *reinterpret_cast<const uint4*>(rp + hi * 16);
      a[mi].q[1] = *reinterpret_cast<const uint4*>(rp + hi * 16 + 8);
    }
    #pragma unroll
    for (int ni = 0; ni < 2; ++ni) {
      const bf16* rp = &Ws[nr + ni * 16 + n16][0];
      b[ni].q[0] = *reinterpret_cast<const uint4*>(rp + hi * 16);
      b[ni].q[1] = *reinterpret_cast<const uint4*>(rp + hi * 16 + 8);
    }
    #pragma unroll
    for (int mi = 0; mi < 2; ++mi)
      #pragma unroll
      for (int ni = 0; ni < 2; ++ni)
        acc[mi][ni] = wmma_bf16(a[mi], b[ni], acc[mi][ni]);
    __syncthreads();
  }

  // store with bias
  #pragma unroll
  for (int mi = 0; mi < 2; ++mi) {
    #pragma unroll
    for (int ni = 0; ni < 2; ++ni) {
      const int ng = n0 + nr + ni * 16 + n16;
      const float bv = bias[ng];
      #pragma unroll
      for (int r = 0; r < 8; ++r) {
        const int mg  = m0 + mr + mi * 16 + r + 8 * hi;
        const float val = acc[mi][ni][r] + bv;
        if (mode == 0) {
          const int b = mg >> 11, s = mg & (SEQ - 1);
          const int h = ng >> 6,  d = ng & (D_HEAD - 1);
          outH[((size_t)(b * N_HEADS + h) * SEQ + s) * D_HEAD + d] = (bf16)val;
        } else {
          outF[(size_t)mg * D_MODEL + ng] = val;
        }
      }
    }
  }
}

// ---------------------------------------------------------------------------
// Two-pass causal attention per (b,h). Block: 128 q-rows (8 waves x 16 rows).
// Pass 1: logits via WMMA, online row max/sum.
// Pass 2: recompute logits, write normalized probs (f32) to d_out weights,
//         accumulate O += P * V via WMMA. Uniform loops -> barriers safe.
// ---------------------------------------------------------------------------
__global__ __launch_bounds__(256)
void attn_kernel(const bf16* __restrict__ Qh, const bf16* __restrict__ Kh,
                 const bf16* __restrict__ Vh, float* __restrict__ Wout,
                 float* __restrict__ Attn) {
  __shared__ bf16 Ks[32][72];       // K tile  [key][d]
  __shared__ bf16 Vt[64][40];       // V tile transposed [d][key]
  __shared__ bf16 Ps[8][16][40];    // per-wave prob tile [m][key]

  const int tid  = threadIdx.x;
  const int wave = tid >> 5, lane = tid & 31;
  const int n16  = lane & 15, hi = lane >> 4;
  const int bh   = blockIdx.y;
  const int Q0   = blockIdx.x * 128 + wave * 16;
  const size_t base = (size_t)bh * SEQ * D_HEAD;
  const float scale = 0.03125f;     // 1/sqrt(1024)

  // Q fragments (16 rows x 64 d) -> two 16x32 A-frags
  FragBF qa0, qa1;
  {
    const bf16* rp = Qh + base + (size_t)(Q0 + n16) * D_HEAD;
    qa0.q[0] = *reinterpret_cast<const uint4*>(rp + hi * 16);
    qa0.q[1] = *reinterpret_cast<const uint4*>(rp + hi * 16 + 8);
    qa1.q[0] = *reinterpret_cast<const uint4*>(rp + 32 + hi * 16);
    qa1.q[1] = *reinterpret_cast<const uint4*>(rp + 32 + hi * 16 + 8);
  }

  float rowmax[8], rowsum[8];
  #pragma unroll
  for (int r = 0; r < 8; ++r) { rowmax[r] = -3.0e38f; rowsum[r] = 0.0f; }

  // ---------------- Pass 1: row statistics ----------------
  const int jsteps1 = (blockIdx.x + 1) * 4;   // keys needed by this block / 32
  for (int jt = 0; jt < jsteps1; ++jt) {
    const int j0 = jt * 32;
    {
      const int row = tid >> 3, c8 = (tid & 7) * 8;
      *reinterpret_cast<uint4*>(&Ks[row][c8]) =
          *reinterpret_cast<const uint4*>(Kh + base + (size_t)(j0 + row) * D_HEAD + c8);
    }
    __syncthreads();
    #pragma unroll
    for (int sn = 0; sn < 2; ++sn) {
      const bf16* kp = &Ks[sn * 16 + n16][0];
      FragBF blo, bhi2;
      blo.q[0]  = *reinterpret_cast<const uint4*>(kp + hi * 16);
      blo.q[1]  = *reinterpret_cast<const uint4*>(kp + hi * 16 + 8);
      bhi2.q[0] = *reinterpret_cast<const uint4*>(kp + 32 + hi * 16);
      bhi2.q[1] = *reinterpret_cast<const uint4*>(kp + 32 + hi * 16 + 8);
      v8f accl = {};
      accl = wmma_bf16(qa0, blo, accl);
      accl = wmma_bf16(qa1, bhi2, accl);
      const int ncol = j0 + sn * 16 + n16;
      #pragma unroll
      for (int r = 0; r < 8; ++r) {
        const int m = Q0 + r + 8 * hi;
        float l = accl[r] * scale + (ncol > m ? -1e9f : 0.0f);
        float tm = l;
        tm = fmaxf(tm, __shfl_xor(tm, 1, 32)); tm = fmaxf(tm, __shfl_xor(tm, 2, 32));
        tm = fmaxf(tm, __shfl_xor(tm, 4, 32)); tm = fmaxf(tm, __shfl_xor(tm, 8, 32));
        const float nm = fmaxf(rowmax[r], tm);
        float ex = __expf(l - nm);
        ex += __shfl_xor(ex, 1, 32); ex += __shfl_xor(ex, 2, 32);
        ex += __shfl_xor(ex, 4, 32); ex += __shfl_xor(ex, 8, 32);
        rowsum[r] = rowsum[r] * __expf(rowmax[r] - nm) + ex;
        rowmax[r] = nm;
      }
    }
    __syncthreads();
  }

  float inv[8];
  #pragma unroll
  for (int r = 0; r < 8; ++r) inv[r] = 1.0f / rowsum[r];

  // ---------------- Pass 2: weights out + P*V ----------------
  v8f accO[4] = {};
  const int qhimax = Q0 + 15;
  for (int jt = 0; jt < SEQ / 32; ++jt) {
    const int j0 = jt * 32;
    {
      const int row = tid >> 3, c8 = (tid & 7) * 8;
      *reinterpret_cast<uint4*>(&Ks[row][c8]) =
          *reinterpret_cast<const uint4*>(Kh + base + (size_t)(j0 + row) * D_HEAD + c8);
      Pack8 pv;
      pv.u = *reinterpret_cast<const uint4*>(Vh + base + (size_t)(j0 + row) * D_HEAD + c8);
      #pragma unroll
      for (int e = 0; e < 8; ++e) Vt[c8 + e][row] = pv.h[e];
    }
    __syncthreads();

    if (j0 <= qhimax) {   // wave-uniform branch
      #pragma unroll
      for (int sn = 0; sn < 2; ++sn) {
        const bf16* kp = &Ks[sn * 16 + n16][0];
        FragBF blo, bhi2;
        blo.q[0]  = *reinterpret_cast<const uint4*>(kp + hi * 16);
        blo.q[1]  = *reinterpret_cast<const uint4*>(kp + hi * 16 + 8);
        bhi2.q[0] = *reinterpret_cast<const uint4*>(kp + 32 + hi * 16);
        bhi2.q[1] = *reinterpret_cast<const uint4*>(kp + 32 + hi * 16 + 8);
        v8f accl = {};
        accl = wmma_bf16(qa0, blo, accl);
        accl = wmma_bf16(qa1, bhi2, accl);
        const int ncol = j0 + sn * 16 + n16;
        #pragma unroll
        for (int r = 0; r < 8; ++r) {
          const int m = Q0 + r + 8 * hi;
          const float l = accl[r] * scale + (ncol > m ? -1e9f : 0.0f);
          const float p = __expf(l - rowmax[r]) * inv[r];
          Wout[((size_t)bh * SEQ + m) * SEQ + ncol] = p;
          Ps[wave][r + 8 * hi][sn * 16 + n16] = (bf16)p;
        }
      }
      FragBF pa;
      const bf16* pp = &Ps[wave][n16][0];
      pa.q[0] = *reinterpret_cast<const uint4*>(pp + hi * 16);
      pa.q[1] = *reinterpret_cast<const uint4*>(pp + hi * 16 + 8);
      #pragma unroll
      for (int nt = 0; nt < 4; ++nt) {
        FragBF vb;
        const bf16* vp = &Vt[nt * 16 + n16][0];
        vb.q[0] = *reinterpret_cast<const uint4*>(vp + hi * 16);
        vb.q[1] = *reinterpret_cast<const uint4*>(vp + hi * 16 + 8);
        accO[nt] = wmma_bf16(pa, vb, accO[nt]);
      }
    } else {
      // fully-masked tile: weights are exactly zero
      #pragma unroll
      for (int sn = 0; sn < 2; ++sn)
        #pragma unroll
        for (int r = 0; r < 8; ++r)
          Wout[((size_t)bh * SEQ + (Q0 + r + 8 * hi)) * SEQ + (j0 + sn * 16 + n16)] = 0.0f;
    }
    __syncthreads();
  }

  // store O (already normalized) into attn buffer [B, S, H*64] as f32
  const int b = bh >> 4, h = bh & 15;
  #pragma unroll
  for (int nt = 0; nt < 4; ++nt) {
    const int d = nt * 16 + n16;
    #pragma unroll
    for (int r = 0; r < 8; ++r) {
      const int s = Q0 + r + 8 * hi;
      Attn[((size_t)(b * SEQ + s)) * D_MODEL + h * D_HEAD + d] = accO[nt][r];
    }
  }
}

// ---------------------------------------------------------------------------
extern "C" void kernel_launch(void* const* d_in, const int* in_sizes, int n_in,
                              void* d_out, int out_size, void* d_ws, size_t ws_size,
                              hipStream_t stream) {
  const float* q  = (const float*)d_in[0];
  const float* k  = (const float*)d_in[1];
  const float* v  = (const float*)d_in[2];
  // d_in[3] = causal mask (recomputed on device, unused)
  const float* Wq = (const float*)d_in[4];
  const float* bq = (const float*)d_in[5];
  const float* Wk = (const float*)d_in[6];
  const float* bk = (const float*)d_in[7];
  const float* Wv = (const float*)d_in[8];
  const float* bv = (const float*)d_in[9];
  const float* Wo = (const float*)d_in[10];
  const float* bo = (const float*)d_in[11];

  float* out  = (float*)d_out;                                   // [B,S,D]
  float* wout = out + (size_t)M_TOK * D_MODEL;                   // [B,H,S,S]

  const size_t headElems = (size_t)BATCH * N_HEADS * SEQ * D_HEAD;  // 8,388,608
  bf16* Qh    = (bf16*)d_ws;
  bf16* Kh    = Qh + headElems;
  bf16* Vh    = Kh + headElems;
  float* AttnF = (float*)(Vh + headElems);                       // [B*S, D] f32

  dim3 gGemm(D_MODEL / 64, M_TOK / 128);      // (16, 64)
  gemm_xwt_kernel<<<gGemm, 256, 0, stream>>>(q, Wq, bq, Qh, nullptr, 0);
  gemm_xwt_kernel<<<gGemm, 256, 0, stream>>>(k, Wk, bk, Kh, nullptr, 0);
  gemm_xwt_kernel<<<gGemm, 256, 0, stream>>>(v, Wv, bv, Vh, nullptr, 0);

  dim3 gAttn(SEQ / 128, BATCH * N_HEADS);     // (16, 64)
  attn_kernel<<<gAttn, 256, 0, stream>>>(Qh, Kh, Vh, wout, AttnF);

  gemm_xwt_kernel<<<gGemm, 256, 0, stream>>>(AttnF, Wo, bo, nullptr, out, 1);
}
